// Attention_24404004176269
// MI455X (gfx1250) — compile-verified, matrix operand order checked by reading, not run on hardware
//
#include <hip/hip_runtime.h>
#include <hip/hip_bf16.h>

typedef __attribute__((ext_vector_type(8)))  float  v8f;
typedef __attribute__((ext_vector_type(16))) __bf16 v16bf;
typedef __attribute__((ext_vector_type(8)))  __bf16 v8bf;

#define N_HEADS 12
#define D_MODEL 768
#define D_HEAD  64
#define N_CTX   2048
#define BATCH   2

// ---------------------------------------------------------------------------
// helpers
// ---------------------------------------------------------------------------
__device__ __forceinline__ v8f zero_v8f() {
    v8f z;
#pragma unroll
    for (int i = 0; i < 8; ++i) z[i] = 0.f;
    return z;
}

__device__ __forceinline__ v8f wmma_bf16(v16bf a, v16bf b, v8f c) {
    // D = A(16x32 bf16) * B(32x16 bf16) + C(16x16 f32)
    return __builtin_amdgcn_wmma_f32_16x16x32_bf16(
        /*neg_a=*/false, a, /*neg_b=*/false, b,
        /*c_mod=*/(short)0, c, /*reuse_a=*/false, /*reuse_b=*/false);
}

// Load a 16-element bf16 A/B fragment for wave32 WMMA from a row-major
// [row][K] buffer.  Per the CDNA5 16-bit fragment layout, lanes 0-15 hold
// K = {k0..k0+7, k0+16..k0+23} and lanes 16-31 hold K = {k0+8..k0+15,
// k0+24..k0+31}; "half" = lane>>4 selects the 8-element sub-chunks.
__device__ __forceinline__ v16bf frag_ld(const __bf16* row, int k0, int half) {
    const __bf16* p0 = row + k0 + 8 * half;
    const __bf16* p1 = p0 + 16;
    v8bf lo = *(const v8bf*)p0;
    v8bf hi = *(const v8bf*)p1;
    v16bf f;
#pragma unroll
    for (int i = 0; i < 8; ++i) { f[i] = lo[i]; f[8 + i] = hi[i]; }
    return f;
}

// xor-swizzle reductions across a 16-lane group (C-fragment N dimension).
// Masks 1,2,4,8 keep lanes 0-15 and 16-31 in separate groups on wave32.
#define SWZ_F32(v, m) \
    __int_as_float(__builtin_amdgcn_ds_swizzle(__float_as_int(v), 0x1F | ((m) << 10)))

__device__ __forceinline__ float red_max16(float v) {
    v = fmaxf(v, SWZ_F32(v, 1));
    v = fmaxf(v, SWZ_F32(v, 2));
    v = fmaxf(v, SWZ_F32(v, 4));
    v = fmaxf(v, SWZ_F32(v, 8));
    return v;
}
__device__ __forceinline__ float red_add16(float v) {
    v += SWZ_F32(v, 1);
    v += SWZ_F32(v, 2);
    v += SWZ_F32(v, 4);
    v += SWZ_F32(v, 8);
    return v;
}

// ---------------------------------------------------------------------------
// Stage 0: weight conversion (f32 -> bf16, transposed for fragment loads)
// ---------------------------------------------------------------------------
// W_{Q,K,V}[h][m][d] (f32)  ->  Wt[h][d][m] (bf16): B-frag rows are d (N), K=m.
__global__ void cvt_wqkv_kernel(const float* __restrict__ W, __bf16* __restrict__ Wt) {
    int idx = blockIdx.x * blockDim.x + threadIdx.x;
    if (idx >= N_HEADS * D_MODEL * D_HEAD) return;
    int d = idx % D_HEAD;
    int m = (idx / D_HEAD) % D_MODEL;
    int h = idx / (D_HEAD * D_MODEL);
    Wt[((size_t)h * D_HEAD + d) * D_MODEL + m] = (__bf16)W[idx];
}

// W_O[h][d][m] (f32) -> Wt[h][m][d] (bf16): B-frag rows are m (N), K=d.
__global__ void cvt_wo_kernel(const float* __restrict__ W, __bf16* __restrict__ Wt) {
    int idx = blockIdx.x * blockDim.x + threadIdx.x;
    if (idx >= N_HEADS * D_HEAD * D_MODEL) return;
    int m = idx % D_MODEL;
    int d = (idx / D_MODEL) % D_HEAD;
    int h = idx / (D_MODEL * D_HEAD);
    Wt[((size_t)h * D_MODEL + m) * D_HEAD + d] = (__bf16)W[idx];
}

// sin/cos rotary tables: [pos][fi], fi in 0..31 (freq repeats for d and d+32)
__global__ void sincos_kernel(float* __restrict__ sin_t, float* __restrict__ cos_t) {
    int idx = blockIdx.x * blockDim.x + threadIdx.x;
    if (idx >= N_CTX * 32) return;
    int fi  = idx & 31;
    int pos = idx >> 5;
    float freq = powf(10000.0f, (float)fi / 32.0f);
    float ang  = (float)pos / freq;
    sin_t[idx] = sinf(ang);
    cos_t[idx] = cosf(ang);
}

// ---------------------------------------------------------------------------
// Stage 1: QKV projection + rotary.  3 waves / block: wave0=Q, wave1=K, wave2=V.
// Each wave computes a 16(seq) x 64(d_head) tile with 24 K-steps of WMMA.
// ---------------------------------------------------------------------------
__global__ void __launch_bounds__(96) qkv_rope_kernel(
    const float* __restrict__ xq, const float* __restrict__ xk, const float* __restrict__ xv,
    const float* __restrict__ bQ, const float* __restrict__ bK, const float* __restrict__ bV,
    const __bf16* __restrict__ WqT, const __bf16* __restrict__ WkT, const __bf16* __restrict__ WvT,
    const float* __restrict__ sin_t, const float* __restrict__ cos_t,
    __bf16* __restrict__ Qbf, __bf16* __restrict__ Kbf, __bf16* __restrict__ Vt)
{
    const int s0   = 16 * blockIdx.x;
    const int h    = blockIdx.y;
    const int b    = blockIdx.z;
    const int lane = threadIdx.x & 31;
    const int wave = threadIdx.x >> 5;
    const int l16  = lane & 15;
    const int half = lane >> 4;

    const float*  x;  const __bf16* WT;  const float* bias;
    if      (wave == 0) { x = xq; WT = WqT; bias = bQ; }
    else if (wave == 1) { x = xk; WT = WkT; bias = bK; }
    else                { x = xv; WT = WvT; bias = bV; }

    // A-fragment source row: x[b][s0+l16][h][:]
    const float*  xrow  = x  + (((size_t)b * N_CTX + (s0 + l16)) * N_HEADS + h) * D_MODEL;
    const __bf16* wbase = WT + (size_t)h * D_HEAD * D_MODEL;

    v8f acc[4];
#pragma unroll
    for (int t = 0; t < 4; ++t) acc[t] = zero_v8f();

    for (int k0 = 0; k0 < D_MODEL; k0 += 32) {
        // A fragment: load f32 activations, convert to bf16 in-register
        v16bf af;
        {
            const v8f lo = *(const v8f*)(xrow + k0 + 8 * half);
            const v8f hi = *(const v8f*)(xrow + k0 + 8 * half + 16);
#pragma unroll
            for (int i = 0; i < 8; ++i) { af[i] = (__bf16)lo[i]; af[8 + i] = (__bf16)hi[i]; }
        }
#pragma unroll
        for (int t = 0; t < 4; ++t) {
            v16bf bf = frag_ld(wbase + (size_t)(16 * t + l16) * D_MODEL, k0, half);
            acc[t] = wmma_bf16(af, bf, acc[t]);
        }
    }

    // bias (broadcast across rows; lane's column is n = 16t + l16)
#pragma unroll
    for (int t = 0; t < 4; ++t) {
        float bv = bias[h * D_HEAD + 16 * t + l16];
#pragma unroll
        for (int r = 0; r < 8; ++r) acc[t][r] += bv;
    }

    if (wave < 2) {
        // rotary: element (row = s0 + r + 8*half, col = 16t + l16).
        // col and col±32 live in the SAME lane (tile t±2) -> lane-local rotate.
        float sv[2][8], cv[2][8];
#pragma unroll
        for (int par = 0; par < 2; ++par) {
            int fi = 16 * par + l16;
#pragma unroll
            for (int r = 0; r < 8; ++r) {
                int pos = s0 + r + 8 * half;
                sv[par][r] = sin_t[pos * 32 + fi];
                cv[par][r] = cos_t[pos * 32 + fi];
            }
        }
        v8f rot[4];
#pragma unroll
        for (int t = 0; t < 4; ++t) {
            int par = t & 1;
#pragma unroll
            for (int r = 0; r < 8; ++r) {
                float flip = (t < 2) ? -acc[t + 2][r] : acc[t - 2][r];
                rot[t][r]  = acc[t][r] * cv[par][r] + flip * sv[par][r];
            }
        }
        __bf16* out = (wave == 0 ? Qbf : Kbf) + (((size_t)b * N_HEADS + h) * N_CTX + s0) * D_HEAD;
#pragma unroll
        for (int t = 0; t < 4; ++t)
#pragma unroll
            for (int r = 0; r < 8; ++r)
                out[(size_t)(r + 8 * half) * D_HEAD + 16 * t + l16] = (__bf16)rot[t][r];
    } else {
        // store V transposed: Vt[b][h][d][s] so P*V B-fragments are contiguous
        __bf16* out = Vt + ((size_t)b * N_HEADS + h) * D_HEAD * N_CTX;
#pragma unroll
        for (int t = 0; t < 4; ++t) {
            int n = 16 * t + l16;
#pragma unroll
            for (int r = 0; r < 8; ++r)
                out[(size_t)n * N_CTX + s0 + 8 * half + r] = (__bf16)acc[t][r];
        }
    }
}

// ---------------------------------------------------------------------------
// Stage 2: causal flash attention.  4 waves / block, each owns one 16-row
// q-tile; kv processed in chunks of 32 (4 score WMMAs + 4 PV WMMAs).
// ---------------------------------------------------------------------------
__global__ void __launch_bounds__(128) attn_kernel(
    const __bf16* __restrict__ Qbf, const __bf16* __restrict__ Kbf,
    const __bf16* __restrict__ Vt,  __bf16* __restrict__ Zbf)
{
    __shared__ __bf16 lds_p[4][16][32];   // per-wave P staging (C-frag -> A-frag)

    const int h    = blockIdx.y;
    const int b    = blockIdx.z;
    const int lane = threadIdx.x & 31;
    const int wave = threadIdx.x >> 5;
    const int l16  = lane & 15;
    const int half = lane >> 4;
    const int q0   = 64 * blockIdx.x + 16 * wave;

    const __bf16* qrow = Qbf + (((size_t)b * N_HEADS + h) * N_CTX + q0 + l16) * D_HEAD;
    const v16bf qf0 = frag_ld(qrow, 0,  half);
    const v16bf qf1 = frag_ld(qrow, 32, half);

    const __bf16* kbase = Kbf + ((size_t)b * N_HEADS + h) * N_CTX * D_HEAD;
    const __bf16* vbase = Vt  + ((size_t)b * N_HEADS + h) * D_HEAD * N_CTX;

    v8f z[4];
    float mrow[8], lrow[8];
#pragma unroll
    for (int t = 0; t < 4; ++t) z[t] = zero_v8f();
#pragma unroll
    for (int r = 0; r < 8; ++r) { mrow[r] = -1e30f; lrow[r] = 0.f; }

    const float scale = 0.125f;  // 1/sqrt(D_HEAD)

    for (int kv0 = 0; kv0 <= q0 + 15; kv0 += 32) {
        // prefetch next K chunk while we compute this one
        if (kv0 + 32 <= q0 + 15)
            __builtin_prefetch(kbase + (size_t)(kv0 + 32 + l16) * D_HEAD, 0, 0);

        // scores: two 16x16 C tiles covering kv columns [kv0, kv0+32)
        v8f sc[2];
#pragma unroll
        for (int tt = 0; tt < 2; ++tt) {
            const __bf16* krow = kbase + (size_t)(kv0 + 16 * tt + l16) * D_HEAD;
            v8f s = zero_v8f();
            s = wmma_bf16(qf0, frag_ld(krow, 0,  half), s);
            s = wmma_bf16(qf1, frag_ld(krow, 32, half), s);
            sc[tt] = s;
        }
        // scale + causal mask (element (row=q0+r+8*half, col=kv0+16tt+l16))
#pragma unroll
        for (int tt = 0; tt < 2; ++tt) {
            int col = kv0 + 16 * tt + l16;
#pragma unroll
            for (int r = 0; r < 8; ++r) {
                float v = sc[tt][r] * scale;
                int row = q0 + r + 8 * half;
                sc[tt][r] = (col > row) ? -1e30f : v;
            }
        }
        // online softmax (per-row state; rows live in vgpr index + half)
#pragma unroll
        for (int r = 0; r < 8; ++r) {
            float mx   = red_max16(fmaxf(sc[0][r], sc[1][r]));
            float mnew = fmaxf(mrow[r], mx);
            float corr = __expf(mrow[r] - mnew);
            float p0   = __expf(sc[0][r] - mnew);
            float p1   = __expf(sc[1][r] - mnew);
            float rs   = red_add16(p0 + p1);
            lrow[r] = lrow[r] * corr + rs;
            mrow[r] = mnew;
#pragma unroll
            for (int t = 0; t < 4; ++t) z[t][r] *= corr;
            // stage P tile in LDS (row-major 16x32, bf16)
            lds_p[wave][r + 8 * half][l16]      = (__bf16)p0;
            lds_p[wave][r + 8 * half][16 + l16] = (__bf16)p1;
        }
        // re-read P as an A fragment (same-wave DS ops are in-order)
        const v16bf pf = frag_ld(&lds_p[wave][l16][0], 0, half);
#pragma unroll
        for (int t = 0; t < 4; ++t) {
            const __bf16* vrow = vbase + (size_t)(16 * t + l16) * N_CTX + kv0;
            z[t] = wmma_bf16(pf, frag_ld(vrow, 0, half), z[t]);
        }
    }

    // normalize and store z as bf16 in [b][s][h][d] for the out-projection
    __bf16* zout = Zbf + (((size_t)b * N_CTX + q0) * N_HEADS + h) * D_HEAD;
#pragma unroll
    for (int t = 0; t < 4; ++t)
#pragma unroll
        for (int r = 0; r < 8; ++r) {
            float v = z[t][r] / lrow[r];
            zout[(size_t)(r + 8 * half) * N_HEADS * D_HEAD + 16 * t + l16] = (__bf16)v;
        }
}

// ---------------------------------------------------------------------------
// Stage 3: per-head output projection z(16x64) @ W_O[h](64x768) + b_O/12.
// 8 waves / block; wave w covers output columns [96w, 96w+96).
// ---------------------------------------------------------------------------
__global__ void __launch_bounds__(256) out_proj_kernel(
    const __bf16* __restrict__ Zbf, const __bf16* __restrict__ WoT,
    const float* __restrict__ bO, float* __restrict__ out)
{
    const int s0   = 16 * blockIdx.x;
    const int h    = blockIdx.y;
    const int b    = blockIdx.z;
    const int lane = threadIdx.x & 31;
    const int wave = threadIdx.x >> 5;
    const int l16  = lane & 15;
    const int half = lane >> 4;
    const int n0   = 96 * wave;

    const __bf16* zrow = Zbf + (((size_t)b * N_CTX + s0 + l16) * N_HEADS + h) * D_HEAD;
    const v16bf za0 = frag_ld(zrow, 0,  half);
    const v16bf za1 = frag_ld(zrow, 32, half);
    const __bf16* wbase = WoT + (size_t)h * D_MODEL * D_HEAD;

    v8f acc[6];
#pragma unroll
    for (int t = 0; t < 6; ++t) acc[t] = zero_v8f();

#pragma unroll
    for (int t = 0; t < 6; ++t) {
        const __bf16* wrow = wbase + (size_t)(n0 + 16 * t + l16) * D_HEAD;
        acc[t] = wmma_bf16(za0, frag_ld(wrow, 0,  half), acc[t]);
        acc[t] = wmma_bf16(za1, frag_ld(wrow, 32, half), acc[t]);
    }

    float* obase = out + (((size_t)b * N_CTX + s0) * N_HEADS + h) * D_MODEL;
#pragma unroll
    for (int t = 0; t < 6; ++t) {
        int   col = n0 + 16 * t + l16;
        float bv  = bO[col] * (1.0f / (float)N_HEADS);
#pragma unroll
        for (int r = 0; r < 8; ++r)
            obase[(size_t)(r + 8 * half) * N_HEADS * D_MODEL + col] = acc[t][r] + bv;
    }
}

// ---------------------------------------------------------------------------
// launch
// ---------------------------------------------------------------------------
extern "C" void kernel_launch(void* const* d_in, const int* in_sizes, int n_in,
                              void* d_out, int out_size, void* d_ws, size_t ws_size,
                              hipStream_t stream)
{
    const float* xq = (const float*)d_in[0];
    const float* xk = (const float*)d_in[1];
    const float* xv = (const float*)d_in[2];
    const float* WQ = (const float*)d_in[3];
    const float* bQ = (const float*)d_in[4];
    const float* WK = (const float*)d_in[5];
    const float* bK = (const float*)d_in[6];
    const float* WV = (const float*)d_in[7];
    const float* bV = (const float*)d_in[8];
    const float* WO = (const float*)d_in[9];
    const float* bO = (const float*)d_in[10];
    float* out = (float*)d_out;

    char* ws = (char*)d_ws;
    size_t off = 0;
    auto take = [&](size_t bytes) -> char* {
        char* p = ws + off;
        off = (off + bytes + 255) & ~(size_t)255;
        return p;
    };
    const size_t wq_bytes  = (size_t)N_HEADS * D_MODEL * D_HEAD * 2;
    const size_t tab_bytes = (size_t)N_CTX * 32 * 4;
    const size_t qkv_bytes = (size_t)BATCH * N_HEADS * N_CTX * D_HEAD * 2;

    __bf16* WqT  = (__bf16*)take(wq_bytes);
    __bf16* WkT  = (__bf16*)take(wq_bytes);
    __bf16* WvT  = (__bf16*)take(wq_bytes);
    __bf16* WoT  = (__bf16*)take(wq_bytes);
    float* sin_t = (float*)take(tab_bytes);
    float* cos_t = (float*)take(tab_bytes);
    __bf16* Qbf  = (__bf16*)take(qkv_bytes);
    __bf16* Kbf  = (__bf16*)take(qkv_bytes);
    __bf16* Vt   = (__bf16*)take(qkv_bytes);
    __bf16* Zbf  = (__bf16*)take(qkv_bytes);
    (void)ws_size; (void)in_sizes; (void)n_in; (void)out_size;

    // Stage 0: weight conversion + rotary tables
    {
        int n = N_HEADS * D_MODEL * D_HEAD;
        int g = (n + 255) / 256;
        cvt_wqkv_kernel<<<g, 256, 0, stream>>>(WQ, WqT);
        cvt_wqkv_kernel<<<g, 256, 0, stream>>>(WK, WkT);
        cvt_wqkv_kernel<<<g, 256, 0, stream>>>(WV, WvT);
        cvt_wo_kernel  <<<g, 256, 0, stream>>>(WO, WoT);
        int nt = N_CTX * 32;
        sincos_kernel<<<(nt + 255) / 256, 256, 0, stream>>>(sin_t, cos_t);
    }
    // Stage 1: QKV projection + rotary
    {
        dim3 grid(N_CTX / 16, N_HEADS, BATCH);
        qkv_rope_kernel<<<grid, 96, 0, stream>>>(xq, xk, xv, bQ, bK, bV,
                                                 WqT, WkT, WvT, sin_t, cos_t,
                                                 Qbf, Kbf, Vt);
    }
    // Stage 2: causal flash attention
    {
        dim3 grid(N_CTX / 64, N_HEADS, BATCH);
        attn_kernel<<<grid, 128, 0, stream>>>(Qbf, Kbf, Vt, Zbf);
    }
    // Stage 3: output projection
    {
        dim3 grid(N_CTX / 16, N_HEADS, BATCH);
        out_proj_kernel<<<grid, 256, 0, stream>>>(Zbf, WoT, bO, out);
    }
}
